// SwinUNETR2D_2284922601645
// MI455X (gfx1250) — matrix operand order repaired and checked by reading, not compiled, not run on hardware
//
#include <hip/hip_runtime.h>
#include <hip/hip_bf16.h>

// ---------------------------------------------------------------------------
// Swin SW-MSA block for gfx1250: all GEMMs via v_wmma_f32_16x16x32_bf16,
// weight panels staged to LDS with double-buffered async copies (ASYNCcnt).
// B=8 H=W=128 C=384 WS=8 SHIFT=4 NH=12 HD=32 N=64
// ---------------------------------------------------------------------------

typedef __attribute__((ext_vector_type(16))) __bf16 v16bf;
typedef __attribute__((ext_vector_type(8)))  float  v8f;
typedef unsigned short u16b;
typedef unsigned int   u32b;

#define DEVINL __device__ __forceinline__

#define TOKENS 131072          // B*H*W
#define CDIM   384
#define QKVDIM 1152
#define FFDIM  1536
#define NHEAD  12
#define HDIM   32
#define SCALE_ATTN 0.17677669529663687f   // 32^-0.5

// ---- gfx1250 async global->LDS copy (guarded; falls back to sync stage) ----
#if defined(__has_builtin)
#if __has_builtin(__builtin_amdgcn_global_load_async_to_lds_b128) && \
    __has_builtin(__builtin_amdgcn_s_wait_asynccnt)
#define USE_ASYNC_COPY 1
#endif
#endif

#ifdef USE_ASYNC_COPY
// Builtin signature (from clang diagnostic): pointee is gcc-vector int4 with
// addrspace(1) source / addrspace(3) destination.
typedef int v4i_ __attribute__((vector_size(16)));
typedef __attribute__((address_space(1))) v4i_ g_v4i;
typedef __attribute__((address_space(3))) v4i_ l_v4i;
#define ASYNC_WAIT(n) __builtin_amdgcn_s_wait_asynccnt(n)
DEVINL void async_cp16(const void* g, void* l) {
  __builtin_amdgcn_global_load_async_to_lds_b128((g_v4i*)g, (l_v4i*)l, 0, 0);
}
#else
#define ASYNC_WAIT(n) ((void)0)
#endif

DEVINL u16b f2bf(float f) {
  u32b u = __float_as_uint(f);
  u += 0x7fffu + ((u >> 16) & 1u);        // round-to-nearest-even
  return (u16b)(u >> 16);
}

union FragU { uint4 q[2]; u16b s[16]; v16bf v; };

// A fragment: 16x32 bf16 tile, row-major source, stride ld (elements).
// Lane l: row = l&15, half = l>>4.  elems 0..15 = K[half*16 .. +15]
// contiguous (matches B fragments).
DEVINL v16bf load_a16x32(const u16b* base, int ld) {
  const int lane = threadIdx.x & 31;
  const int m = lane & 15, hf = lane >> 4;
  FragU u;
  const u16b* p = base + (size_t)m * ld + hf * 16;
  u.q[0] = *reinterpret_cast<const uint4*>(p);
  u.q[1] = *reinterpret_cast<const uint4*>(p + 8);
  return u.v;
}

// B fragment from B^T stored row-major (N x K), stride ld.
// Lane l: col n = l&15, elems 0..15 = K[half*16 .. +15] contiguous.
DEVINL v16bf load_b_nk(const u16b* base, int ld) {
  const int lane = threadIdx.x & 31;
  const int n = lane & 15, hf = lane >> 4;
  FragU u;
  const u16b* p = base + (size_t)n * ld + hf * 16;
  u.q[0] = *reinterpret_cast<const uint4*>(p);
  u.q[1] = *reinterpret_cast<const uint4*>(p + 8);
  return u.v;
}

// B fragment from B stored row-major (K x N), stride ld (strided gather).
DEVINL v16bf load_b_kn(const u16b* base, int ld) {
  const int lane = threadIdx.x & 31;
  const int n = lane & 15, hf = lane >> 4;
  FragU u;
#pragma unroll
  for (int j = 0; j < 16; ++j) u.s[j] = base[(size_t)(hf * 16 + j) * ld + n];
  return u.v;
}

DEVINL v8f wmma_bf16(v16bf a, v16bf b, v8f c) {
  return __builtin_amdgcn_wmma_f32_16x16x32_bf16(
      /*neg_a=*/false, a, /*neg_b=*/false, b,
      /*c_mod=*/(short)0, c, /*reuse_a=*/false, /*reuse_b=*/false);
}

// Stage one 64(n) x 32(k) bf16 weight chunk (4KB) into LDS, [n][32] layout.
// 128 threads x 32B each.
DEVINL void stage_b64x32(const u16b* __restrict__ W, int ldw, u16b* ldsDst) {
  const int tid = threadIdx.x;
  const int n = tid >> 1;                // 0..63
  const int kk = (tid & 1) * 16;         // 0 or 16
  const u16b* src = W + (size_t)n * ldw + kk;
  u16b* dst = ldsDst + n * 32 + kk;
#ifdef USE_ASYNC_COPY
  async_cp16(src, dst);
  async_cp16(src + 8, dst + 8);
#else
  *(uint4*)dst = *(const uint4*)src;
  *(uint4*)(dst + 8) = *(const uint4*)(src + 8);
#endif
}

// 64x64 output tile: 4 waves/block, wave owns 16 rows x 64 cols.
// B chunks double-buffered in LDS; async DMA for step i+1 overlaps WMMA of i.
// Fully unrolled so accumulators stay pinned (no phi-copy mov blocks).
template<int KDIM>
DEVINL void gemm64x64(const u16b* __restrict__ A, const u16b* __restrict__ W,
                      int row0, int col0, u16b* ldsB /* [2][64*32] */,
                      v8f acc[4]) {
  constexpr int NSTEP = KDIM / 32;
  const u16b* a_base = A + (size_t)row0 * KDIM;
  const u16b* w_base = W + (size_t)col0 * KDIM;

  stage_b64x32(w_base, KDIM, ldsB);                       // step 0 -> buf 0
#pragma unroll
  for (int i = 0; i < NSTEP - 1; ++i) {
    stage_b64x32(w_base + (i + 1) * 32, KDIM, ldsB + ((i + 1) & 1) * 2048);
    ASYNC_WAIT(2);                       // own step-i transfers complete
    __syncthreads();                     // everyone's buf_cur data visible
    const u16b* bb = ldsB + (i & 1) * 2048;
    if (i + 1 < NSTEP) __builtin_prefetch(a_base + (i + 1) * 32, 0, 1);
    v16bf a = load_a16x32(a_base + i * 32, KDIM);
#pragma unroll
    for (int t = 0; t < 4; ++t) {
      v16bf b = load_b_nk(bb + t * 16 * 32, 32);
      acc[t] = wmma_bf16(a, b, acc[t]);
    }
    __syncthreads();                     // reads done before buffer reuse
  }
  ASYNC_WAIT(0);
  __syncthreads();
  {
    const int i = NSTEP - 1;
    const u16b* bb = ldsB + (i & 1) * 2048;
    v16bf a = load_a16x32(a_base + i * 32, KDIM);
#pragma unroll
    for (int t = 0; t < 4; ++t) {
      v16bf b = load_b_nk(bb + t * 16 * 32, 32);
      acc[t] = wmma_bf16(a, b, acc[t]);
    }
  }
}

// ---------------------------------------------------------------------------
__global__ void __launch_bounds__(256) kcvt_bf16(const float* __restrict__ s,
                                                 u16b* __restrict__ d, int n) {
  int i = blockIdx.x * blockDim.x + threadIdx.x;
  int st = gridDim.x * blockDim.x;
  for (; i < n; i += st) d[i] = f2bf(s[i]);
}

// LN1 + cyclic shift (-4,-4) + window partition -> bf16 [2048][64][384]
__global__ void __launch_bounds__(256) kln1_shift(const float* __restrict__ x,
                                                  const float* __restrict__ g,
                                                  const float* __restrict__ bt,
                                                  u16b* __restrict__ xw) {
  const int wid = threadIdx.x >> 5, lane = threadIdx.x & 31;
  const int t = blockIdx.x * 8 + wid;                 // token id, 0..131071
  const float* xp = x + (size_t)t * CDIM;
  float xv[12], s = 0.f, s2 = 0.f;
#pragma unroll
  for (int q = 0; q < 12; ++q) { float v = xp[lane + 32 * q]; xv[q] = v; s += v; s2 += v * v; }
#pragma unroll
  for (int m = 1; m < 32; m <<= 1) { s += __shfl_xor(s, m, 32); s2 += __shfl_xor(s2, m, 32); }
  const float mu = s * (1.f / CDIM);
  const float rs = rsqrtf(s2 * (1.f / CDIM) - mu * mu + 1e-5f);
  const int b = t >> 14, rem = t & 16383, i = rem >> 7, j = rem & 127;
  const int is = (i + 124) & 127, js = (j + 124) & 127;   // roll by -SHIFT
  const int w = (b << 8) + ((is >> 3) << 4) + (js >> 3);
  const int pos = ((is & 7) << 3) + (js & 7);
  u16b* dp = xw + ((size_t)w * 64 + pos) * CDIM;
#pragma unroll
  for (int q = 0; q < 12; ++q) {
    int c = lane + 32 * q;
    dp[c] = f2bf((xv[q] - mu) * rs * g[c] + bt[c]);
  }
}

// LN2 (linear layout) -> bf16
__global__ void __launch_bounds__(256) kln2(const float* __restrict__ x,
                                            const float* __restrict__ g,
                                            const float* __restrict__ bt,
                                            u16b* __restrict__ xo) {
  const int wid = threadIdx.x >> 5, lane = threadIdx.x & 31;
  const int t = blockIdx.x * 8 + wid;
  const float* xp = x + (size_t)t * CDIM;
  float xv[12], s = 0.f, s2 = 0.f;
#pragma unroll
  for (int q = 0; q < 12; ++q) { float v = xp[lane + 32 * q]; xv[q] = v; s += v; s2 += v * v; }
#pragma unroll
  for (int m = 1; m < 32; m <<= 1) { s += __shfl_xor(s, m, 32); s2 += __shfl_xor(s2, m, 32); }
  const float mu = s * (1.f / CDIM);
  const float rs = rsqrtf(s2 * (1.f / CDIM) - mu * mu + 1e-5f);
  u16b* dp = xo + (size_t)t * CDIM;
#pragma unroll
  for (int q = 0; q < 12; ++q) {
    int c = lane + 32 * q;
    dp[c] = f2bf((xv[q] - mu) * rs * g[c] + bt[c]);
  }
}

// QKV: [131072,384]bf16 x W(1152x384) -> bf16 [tok][1152]
__global__ void __launch_bounds__(128) kqkv(const u16b* __restrict__ A,
                                            const u16b* __restrict__ W,
                                            const float* __restrict__ bias,
                                            u16b* __restrict__ out) {
  __shared__ __align__(16) u16b ldsB[2 * 2048];
  const int wid = threadIdx.x >> 5, lane = threadIdx.x & 31;
  const int nloc = lane & 15, hf = lane >> 4;
  const int row0 = blockIdx.y * 64 + wid * 16;
  const int col0 = blockIdx.x * 64;
  v8f acc[4] = {};
  gemm64x64<CDIM>(A, W, row0, col0, ldsB, acc);
#pragma unroll
  for (int t = 0; t < 4; ++t) {
    const int col = col0 + 16 * t + nloc;
    const float bv = bias[col];
#pragma unroll
    for (int r = 0; r < 8; ++r) {
      const int row = row0 + hf * 8 + r;
      out[(size_t)row * QKVDIM + col] = f2bf(acc[t][r] + bv);
    }
  }
}

// Attention: one block (4 waves) per (window, head)
__global__ void __launch_bounds__(128) kattn(const u16b* __restrict__ qkv,
                                             const float* __restrict__ relTab,
                                             const float* __restrict__ mask,
                                             u16b* __restrict__ aout) {
  __shared__ __align__(16) u16b ldsP[64 * 64];        // softmax probs, bf16
  const int wid = threadIdx.x >> 5, lane = threadIdx.x & 31;
  const int nloc = lane & 15, hf = lane >> 4;
  const int h = blockIdx.x;                            // head 0..11
  const int w = blockIdx.y;                            // window 0..2047
  const size_t base = (size_t)w * 64 * QKVDIM;
  const u16b* Q = qkv + base + h * HDIM;
  const u16b* K = qkv + base + CDIM + h * HDIM;
  const u16b* V = qkv + base + 2 * CDIM + h * HDIM;

  // S = Q K^T  (wave rows wid*16..+15, all 64 cols; K=32 -> one step)
  v8f S[4] = {};
  {
    v16bf a = load_a16x32(Q + (size_t)(wid * 16) * QKVDIM, QKVDIM);
#pragma unroll
    for (int t = 0; t < 4; ++t) {
      v16bf b = load_b_nk(K + (size_t)(t * 16) * QKVDIM, QKVDIM);
      S[t] = wmma_bf16(a, b, S[t]);
    }
  }
  // scale + relative position bias (index computed on the fly) + shift mask
  const float* mrow = mask + (size_t)(w & 255) * 4096;
  float sv[4][8];
#pragma unroll
  for (int t = 0; t < 4; ++t) {
    const int c = t * 16 + nloc;
    const int chh = c >> 3, cww = c & 7;
#pragma unroll
    for (int r = 0; r < 8; ++r) {
      const int m = wid * 16 + hf * 8 + r;
      const int ridx = ((m >> 3) - chh + 7) * 15 + ((m & 7) - cww + 7);
      sv[t][r] = S[t][r] * SCALE_ATTN + relTab[ridx * NHEAD + h] + mrow[m * 64 + c];
    }
  }
  // softmax per row: each row lives in one 16-lane half-group
#pragma unroll
  for (int r = 0; r < 8; ++r) {
    float mx = fmaxf(fmaxf(sv[0][r], sv[1][r]), fmaxf(sv[2][r], sv[3][r]));
#pragma unroll
    for (int d = 1; d < 16; d <<= 1) mx = fmaxf(mx, __shfl_xor(mx, d, 32));
    float sum = 0.f;
#pragma unroll
    for (int t = 0; t < 4; ++t) { sv[t][r] = __expf(sv[t][r] - mx); sum += sv[t][r]; }
#pragma unroll
    for (int d = 1; d < 16; d <<= 1) sum += __shfl_xor(sum, d, 32);
    const float inv = 1.0f / sum;
#pragma unroll
    for (int t = 0; t < 4; ++t) sv[t][r] *= inv;
  }
  // stage P in LDS to re-read in A-fragment layout
#pragma unroll
  for (int t = 0; t < 4; ++t) {
    const int c = t * 16 + nloc;
#pragma unroll
    for (int r = 0; r < 8; ++r) {
      const int m = wid * 16 + hf * 8 + r;
      ldsP[m * 64 + c] = f2bf(sv[t][r]);
    }
  }
  __syncthreads();
  // O = P V  (16x64 x 64x32 per wave)
  v8f O[2] = {};
#pragma unroll
  for (int ks = 0; ks < 2; ++ks) {
    v16bf a = load_a16x32(ldsP + (size_t)(wid * 16) * 64 + ks * 32, 64);
#pragma unroll
    for (int t = 0; t < 2; ++t) {
      v16bf b = load_b_kn(V + (size_t)(ks * 32) * QKVDIM + t * 16, QKVDIM);
      O[t] = wmma_bf16(a, b, O[t]);
    }
  }
#pragma unroll
  for (int t = 0; t < 2; ++t) {
#pragma unroll
    for (int r = 0; r < 8; ++r) {
      const int m = wid * 16 + hf * 8 + r;
      aout[((size_t)w * 64 + m) * CDIM + h * HDIM + t * 16 + nloc] = f2bf(O[t][r]);
    }
  }
}

// Proj + window-reverse + un-shift + residual -> f32 x1 (linear token layout)
__global__ void __launch_bounds__(128) kproj(const u16b* __restrict__ A,
                                             const u16b* __restrict__ W,
                                             const float* __restrict__ bias,
                                             const float* __restrict__ xres,
                                             float* __restrict__ x1) {
  __shared__ __align__(16) u16b ldsB[2 * 2048];
  const int wid = threadIdx.x >> 5, lane = threadIdx.x & 31;
  const int nloc = lane & 15, hf = lane >> 4;
  const int row0 = blockIdx.y * 64 + wid * 16;
  const int col0 = blockIdx.x * 64;
  v8f acc[4] = {};
  gemm64x64<CDIM>(A, W, row0, col0, ldsB, acc);
#pragma unroll
  for (int t = 0; t < 4; ++t) {
    const int col = col0 + 16 * t + nloc;
    const float bv = bias[col];
#pragma unroll
    for (int r = 0; r < 8; ++r) {
      const int tw = row0 + hf * 8 + r;         // windowed token id
      const int w = tw >> 6, pos = tw & 63;
      const int b = w >> 8, win = w & 255;
      const int i_sh = ((win >> 4) << 3) + (pos >> 3);
      const int j_sh = ((win & 15) << 3) + (pos & 7);
      const int i = (i_sh + 4) & 127, j = (j_sh + 4) & 127;   // roll +SHIFT
      const size_t tok = ((size_t)b << 14) + (i << 7) + j;
      x1[tok * CDIM + col] = xres[tok * CDIM + col] + acc[t][r] + bv;
    }
  }
}

// FC1 + exact GELU -> bf16
__global__ void __launch_bounds__(128) kfc1(const u16b* __restrict__ A,
                                            const u16b* __restrict__ W,
                                            const float* __restrict__ bias,
                                            u16b* __restrict__ out) {
  __shared__ __align__(16) u16b ldsB[2 * 2048];
  const int wid = threadIdx.x >> 5, lane = threadIdx.x & 31;
  const int nloc = lane & 15, hf = lane >> 4;
  const int row0 = blockIdx.y * 64 + wid * 16;
  const int col0 = blockIdx.x * 64;
  v8f acc[4] = {};
  gemm64x64<CDIM>(A, W, row0, col0, ldsB, acc);
#pragma unroll
  for (int t = 0; t < 4; ++t) {
    const int col = col0 + 16 * t + nloc;
    const float bv = bias[col];
#pragma unroll
    for (int r = 0; r < 8; ++r) {
      const int row = row0 + hf * 8 + r;
      float v = acc[t][r] + bv;
      v = 0.5f * v * (1.0f + erff(v * 0.70710678118654752f));
      out[(size_t)row * FFDIM + col] = f2bf(v);
    }
  }
}

// FC2 + residual -> f32 final output
__global__ void __launch_bounds__(128) kfc2(const u16b* __restrict__ A,
                                            const u16b* __restrict__ W,
                                            const float* __restrict__ bias,
                                            const float* __restrict__ x1,
                                            float* __restrict__ out) {
  __shared__ __align__(16) u16b ldsB[2 * 2048];
  const int wid = threadIdx.x >> 5, lane = threadIdx.x & 31;
  const int nloc = lane & 15, hf = lane >> 4;
  const int row0 = blockIdx.y * 64 + wid * 16;
  const int col0 = blockIdx.x * 64;
  v8f acc[4] = {};
  gemm64x64<FFDIM>(A, W, row0, col0, ldsB, acc);
#pragma unroll
  for (int t = 0; t < 4; ++t) {
    const int col = col0 + 16 * t + nloc;
    const float bv = bias[col];
#pragma unroll
    for (int r = 0; r < 8; ++r) {
      const int row = row0 + hf * 8 + r;
      out[(size_t)row * CDIM + col] = x1[(size_t)row * CDIM + col] + acc[t][r] + bv;
    }
  }
}

// ---------------------------------------------------------------------------
extern "C" void kernel_launch(void* const* d_in, const int* in_sizes, int n_in,
                              void* d_out, int out_size, void* d_ws, size_t ws_size,
                              hipStream_t stream) {
  (void)in_sizes; (void)n_in; (void)out_size; (void)ws_size;
  const float* x       = (const float*)d_in[0];
  const float* qkv_w   = (const float*)d_in[1];
  const float* qkv_b   = (const float*)d_in[2];
  const float* proj_w  = (const float*)d_in[3];
  const float* proj_b  = (const float*)d_in[4];
  const float* rel_tab = (const float*)d_in[5];
  const float* g1      = (const float*)d_in[6];
  const float* b1      = (const float*)d_in[7];
  const float* g2      = (const float*)d_in[8];
  const float* b2      = (const float*)d_in[9];
  const float* fc1_w   = (const float*)d_in[10];
  const float* fc1_b   = (const float*)d_in[11];
  const float* fc2_w   = (const float*)d_in[12];
  const float* fc2_b   = (const float*)d_in[13];
  const float* amask   = (const float*)d_in[14];
  float* out = (float*)d_out;

  char* ws = (char*)d_ws;
  size_t off = 0;
  auto a256 = [](size_t v) { return (v + 255) & ~(size_t)255; };

  u16b* w_qkv  = (u16b*)(ws + off); off = a256(off + (size_t)QKVDIM * CDIM * 2);
  u16b* w_proj = (u16b*)(ws + off); off = a256(off + (size_t)CDIM * CDIM * 2);
  u16b* w_fc1  = (u16b*)(ws + off); off = a256(off + (size_t)FFDIM * CDIM * 2);
  u16b* w_fc2  = (u16b*)(ws + off); off = a256(off + (size_t)CDIM * FFDIM * 2);
  u16b* xw     = (u16b*)(ws + off); off = a256(off + (size_t)TOKENS * CDIM * 2);  // also x1_ln
  u16b* big    = (u16b*)(ws + off); off = a256(off + (size_t)TOKENS * FFDIM * 2); // qkv|attn / fc1
  float* x1    = (float*)(ws + off); off = a256(off + (size_t)TOKENS * CDIM * 4);

  u16b* qkvbuf = big;
  u16b* attn_o = big + (size_t)TOKENS * QKVDIM;   // 1152+384 == 1536: fits exactly
  u16b* fc1buf = big;                             // reused after attention is done
  u16b* x1_ln  = xw;                              // reused after QKV GEMM

  // 0) weights -> bf16
  kcvt_bf16<<<512, 256, 0, stream>>>(qkv_w,  w_qkv,  QKVDIM * CDIM);
  kcvt_bf16<<<512, 256, 0, stream>>>(proj_w, w_proj, CDIM * CDIM);
  kcvt_bf16<<<512, 256, 0, stream>>>(fc1_w,  w_fc1,  FFDIM * CDIM);
  kcvt_bf16<<<512, 256, 0, stream>>>(fc2_w,  w_fc2,  CDIM * FFDIM);

  // 1) LN1 + shift + window partition
  kln1_shift<<<TOKENS / 8, 256, 0, stream>>>(x, g1, b1, xw);

  // 2) QKV GEMM  (M=131072, N=1152, K=384)
  kqkv<<<dim3(QKVDIM / 64, TOKENS / 64), 128, 0, stream>>>(xw, w_qkv, qkv_b, qkvbuf);

  // 3) windowed attention (2048 windows x 12 heads)
  kattn<<<dim3(NHEAD, TOKENS / 64), 128, 0, stream>>>(qkvbuf, rel_tab, amask, attn_o);

  // 4) proj + window reverse + residual
  kproj<<<dim3(CDIM / 64, TOKENS / 64), 128, 0, stream>>>(attn_o, w_proj, proj_b, x, x1);

  // 5) LN2
  kln2<<<TOKENS / 8, 256, 0, stream>>>(x1, g2, b2, x1_ln);

  // 6) FC1 + GELU  (N=1536)
  kfc1<<<dim3(FFDIM / 64, TOKENS / 64), 128, 0, stream>>>(x1_ln, w_fc1, fc1_b, fc1buf);

  // 7) FC2 + residual  (K=1536)
  kfc2<<<dim3(CDIM / 64, TOKENS / 64), 128, 0, stream>>>(fc1buf, w_fc2, fc2_b, x1, out);
}
